// TernaryMobileNet_16363825398000
// MI455X (gfx1250) — matrix operand order repaired and checked by reading, not compiled, not run on hardware
//
#include <hip/hip_runtime.h>
#include <hip/hip_bf16.h>

typedef __attribute__((ext_vector_type(16))) _Float16 v16h;
typedef __attribute__((ext_vector_type(8)))  _Float16 v8h;
typedef __attribute__((ext_vector_type(8)))  float    v8f;
typedef __attribute__((ext_vector_type(4)))  int      v4i;

#define EPSV 1e-5f

#if __has_builtin(__builtin_amdgcn_global_load_async_to_lds_b128) && \
    __has_builtin(__builtin_amdgcn_s_wait_asynccnt)
#define USE_ASYNC_LDS 1
#else
#define USE_ASYNC_LDS 0
#endif

// ---------------------------------------------------------------------------
// Ternary (TWN) stats: delta = 0.7*mean|w|, alpha = mean|w| over |w|>delta.
// One block per weight tensor (w1, w2, w3, wl).
// ---------------------------------------------------------------------------
__global__ __launch_bounds__(256) void ternary_stats_kernel(
    const float* __restrict__ w0, const float* __restrict__ w1,
    const float* __restrict__ w2, const float* __restrict__ w3,
    int n0, int n1, int n2, int n3, float* __restrict__ stats)
{
  __shared__ float red[256];
  __shared__ float sdelta;
  const float* w; int n;
  if      (blockIdx.x == 0) { w = w0; n = n0; }
  else if (blockIdx.x == 1) { w = w1; n = n1; }
  else if (blockIdx.x == 2) { w = w2; n = n2; }
  else                      { w = w3; n = n3; }
  const int tid = threadIdx.x;

  float s = 0.f;
  for (int i = tid; i < n; i += 256) s += fabsf(w[i]);
  red[tid] = s; __syncthreads();
  for (int off = 128; off > 0; off >>= 1) {
    if (tid < off) red[tid] += red[tid + off];
    __syncthreads();
  }
  if (tid == 0) sdelta = 0.7f * red[0] / (float)n;
  __syncthreads();
  const float delta = sdelta;

  float ms = 0.f, cnt = 0.f;
  for (int i = tid; i < n; i += 256) {
    float a = fabsf(w[i]);
    if (a > delta) { ms += a; cnt += 1.f; }
  }
  __syncthreads();
  red[tid] = ms; __syncthreads();
  for (int off = 128; off > 0; off >>= 1) {
    if (tid < off) red[tid] += red[tid + off];
    __syncthreads();
  }
  const float mstot = red[0];
  __syncthreads();
  red[tid] = cnt; __syncthreads();
  for (int off = 128; off > 0; off >>= 1) {
    if (tid < off) red[tid] += red[tid + off];
    __syncthreads();
  }
  if (tid == 0) {
    stats[blockIdx.x * 2 + 0] = delta;
    stats[blockIdx.x * 2 + 1] = mstot / fmaxf(red[0], 1.f);
  }
}

// ---------------------------------------------------------------------------
// Pack ternarized weights as f16 with BN scale gamma/sqrt(var+eps) folded in.
// POSITION-MAJOR K layout:  k = (kh*3+kw)*CIN + ci.
// wpack[o][k], k padded to KPAD (multiple of 32).
// ---------------------------------------------------------------------------
__global__ __launch_bounds__(256) void pack_weights_kernel(
    const float* __restrict__ w, const float* __restrict__ gamma,
    const float* __restrict__ var, const float* __restrict__ stats,
    _Float16* __restrict__ wpack, int Cout, int Cin, int Kpad)
{
  int idx = blockIdx.x * blockDim.x + threadIdx.x;
  int total = Cout * Kpad;
  if (idx >= total) return;
  int o = idx / Kpad;
  int k = idx - o * Kpad;
  int Kreal = Cin * 9;
  float val = 0.f;
  if (k < Kreal) {
    int pos = k / Cin;            // kh*3+kw
    int ci  = k - pos * Cin;
    float delta = stats[0], alpha = stats[1];
    float wv = w[(o * Cin + ci) * 9 + pos];   // original OIHW order
    float aw = fabsf(wv);
    float q = (aw > delta) ? ((wv > 0.f) ? alpha : -alpha) : 0.f;
    float inv = gamma[o] * rsqrtf(var[o] + EPSV);
    val = q * inv;
  }
  wpack[idx] = (_Float16)val;
}

// ---------------------------------------------------------------------------
// Implicit-GEMM conv 3x3 pad 1 via v_wmma_f32_16x16x32_f16.
// Activations NHWC f16 (first layer reads NCHW f32 via generic path).
// One M-tile per wave, ALL NT = COUT/16 N-tiles per wave (A reused NT times).
// Fast path (CIN%32==0):
//  * whole B tile staged to LDS once per block, FRAGMENT-MAJOR
//    [kstep][nt][lane][16 halfs] -> inner-loop B read = 32B contiguous/lane
//    (async global->LDS when the toolchain exposes the builtins)
//  * lane's 16 A halfs = channels {cb..cb+7, cb+16..cb+23} at fixed (ih,iw)
//    -> two aligned global b128 loads.
// ---------------------------------------------------------------------------
template<typename TIN, int CIN, int COUT, int IH, int IW, int OH, int OW, int STRIDE>
__global__ __launch_bounds__(128) void conv_wmma_kernel(
    const TIN* __restrict__ in, const _Float16* __restrict__ wpack,
    const float* __restrict__ gamma, const float* __restrict__ beta,
    const float* __restrict__ mean,  const float* __restrict__ var,
    _Float16* __restrict__ out)
{
  constexpr int KREAL = CIN * 9;
  constexpr int KPAD  = ((KREAL + 31) / 32) * 32;
  constexpr int NT    = COUT / 16;
  constexpr int IHW   = IH * IW;
  constexpr int OHW   = OH * OW;
  constexpr int KSTEPS = KPAD / 32;

  extern __shared__ _Float16 smem[];   // [KSTEPS][NT][32][16] fragment-major

  const int lane   = threadIdx.x & 31;
  const int wave   = threadIdx.x >> 5;
  const int lane15 = lane & 15;
  const int hi     = (lane & 16) ? 1 : 0;     // half-wave selector
  const int tileM  = (blockIdx.x * 4 + wave) * 16;

  // ---- stage B tile into LDS (fast path only), one copy per block ----
  if constexpr (CIN % 32 == 0) {
    constexpr int NCHUNK = KSTEPS * NT * 32;  // 32B chunks
    for (int c = threadIdx.x; c < NCHUNK; c += 128) {
      const int ksIdx = c / (NT * 32);
      const int r     = c - ksIdx * (NT * 32);
      const int nt    = r >> 5;
      const int ln    = r & 31;
      const _Float16* src = wpack + (size_t)(nt * 16 + (ln & 15)) * KPAD
                                  + ksIdx * 32 + ((ln & 16) ? 8 : 0);
      _Float16* dst = smem + (size_t)c * 16;
#if USE_ASYNC_LDS
      __builtin_amdgcn_global_load_async_to_lds_b128(
          (__attribute__((address_space(1))) v4i*)src,
          (__attribute__((address_space(3))) v4i*)dst, 0, 0);
      __builtin_amdgcn_global_load_async_to_lds_b128(
          (__attribute__((address_space(1))) v4i*)(src + 16),
          (__attribute__((address_space(3))) v4i*)(dst + 8), 0, 0);
#else
      *(v8h*)dst       = *(const v8h*)src;
      *(v8h*)(dst + 8) = *(const v8h*)(src + 16);
#endif
    }
#if USE_ASYNC_LDS
    __builtin_amdgcn_s_wait_asynccnt(0);
#endif
    __syncthreads();
  }

  // one decode per wave-tile: (n, oh0, ow0); lanes offset ow only
  const int n    = tileM / OHW;
  const int rem  = tileM - n * OHW;
  const int oh0  = rem / OW;
  const int ow0  = rem - oh0 * OW;
  const int owl  = ow0 + lane15;              // this lane's A-row ow

  // per-N-tile BN bias (scale folded into wpack)
  float bias[NT];
#pragma unroll
  for (int nt = 0; nt < NT; ++nt) {
    int ch = nt * 16 + lane15;
    float inv = gamma[ch] * rsqrtf(var[ch] + EPSV);
    bias[nt] = beta[ch] - mean[ch] * inv;
  }

  const int kA = hi ? 8 : 0;                  // A-fragment K sub-offset
  const _Float16* wbase = wpack + (size_t)lane15 * KPAD + (hi ? 8 : 0);

  v8f acc[NT];
#pragma unroll
  for (int nt = 0; nt < NT; ++nt) acc[nt] = (v8f){};

  for (int ks = 0; ks < KPAD; ks += 32) {
    // ---- assemble A fragment (16 halfs / lane) ----
    v16h a;
    if constexpr (CIN % 32 == 0) {
      const int pos = ks / CIN;                     // filter tap (uniform)
      const int cb  = (ks - pos * CIN) + kA;        // channel base (mult of 8)
      const int kh  = pos / 3, kw = pos - kh * 3;
      const int ih  = oh0 * STRIDE - 1 + kh;
      const int iw  = owl * STRIDE - 1 + kw;
      const bool inb = ((unsigned)ih < (unsigned)IH) & ((unsigned)iw < (unsigned)IW);
      // NHWC: channels contiguous at fixed (n, ih, iw)
      const TIN* ab = in + (((size_t)n * IH + ih) * IW + iw) * CIN + cb;
      __builtin_prefetch(ab + (size_t)IW * CIN, 0, 0);   // next input row
      v8h lo = inb ? *(const v8h*)ab        : (v8h){};
      v8h hv = inb ? *(const v8h*)(ab + 16) : (v8h){};
#pragma unroll
      for (int i = 0; i < 8; ++i) { a[i] = lo[i]; a[i + 8] = hv[i]; }
    } else {
      // generic path (layer 1: NCHW f32 input, CIN=3, K=27, single K-step)
      __builtin_prefetch(wbase + ks + 32, 0, 0);
#pragma unroll
      for (int i = 0; i < 16; ++i) {
        const int kk = ks + kA + ((i < 8) ? i : i + 8);
        float av = 0.f;
        if (kk < KREAL) {
          int pos = kk / CIN;
          int ci  = kk - pos * CIN;
          int kh  = pos / 3, kw = pos - kh * 3;
          int ih  = oh0 * STRIDE - 1 + kh;
          int iw  = owl * STRIDE - 1 + kw;
          if ((unsigned)ih < (unsigned)IH && (unsigned)iw < (unsigned)IW)
            av = (float)in[((size_t)n * CIN + ci) * IHW + (size_t)ih * IW + iw];
        }
        a[i] = (_Float16)av;
      }
    }
    // ---- NT WMMAs reuse this A fragment ----
#pragma unroll
    for (int nt = 0; nt < NT; ++nt) {
      v16h b;
      if constexpr (CIN % 32 == 0) {
        // LDS, fragment-major: lane's own contiguous 32B chunk
        b = *(const v16h*)(smem + (size_t)((((ks >> 5) * NT + nt) * 32) + lane) * 16);
      } else {
        v8h b0 = *(const v8h*)(wbase + (size_t)nt * 16 * KPAD + ks);
        v8h b1 = *(const v8h*)(wbase + (size_t)nt * 16 * KPAD + ks + 16);
#pragma unroll
        for (int i = 0; i < 8; ++i) { b[i] = b0[i]; b[i + 8] = b1[i]; }
      }
      acc[nt] = __builtin_amdgcn_wmma_f32_16x16x32_f16(
          false, a, false, b, (short)0, acc[nt], false, false);
    }
  }

  // ---- epilogue: BN bias + ReLU, NHWC stores ----
  // C/D layout: lane L, vgpr r -> (M = r + 8*hi, N = L%16); M maps to ow0+M.
#pragma unroll
  for (int nt = 0; nt < NT; ++nt) {
    const int ch = nt * 16 + lane15;
    _Float16* op = out + (((size_t)n * OH + oh0) * OW + ow0 + 8 * hi) * COUT + ch;
#pragma unroll
    for (int r = 0; r < 8; ++r)
      op[(size_t)r * COUT] = (_Float16)fmaxf(acc[nt][r] + bias[nt], 0.f);
  }
}

// ---------------------------------------------------------------------------
// Global average pool over NHWC h3 [64][56*56][128]: one block per image,
// one thread per channel; spatial loop is fully coalesced across threads.
// ---------------------------------------------------------------------------
__global__ __launch_bounds__(128) void avgpool_nhwc_kernel(
    const _Float16* __restrict__ h, float* __restrict__ pooled)
{
  const int n = blockIdx.x;
  const int c = threadIdx.x;        // 128 channels
  const _Float16* p = h + (size_t)n * 3136 * 128 + c;
  float s = 0.f;
  for (int i = 0; i < 3136; ++i) s += (float)p[(size_t)i * 128];
  pooled[n * 128 + c] = s * (1.f / 3136.f);
}

// ---------------------------------------------------------------------------
// Linear head: out[n][o] = pooled[n] . ternarize(wl[o]) + bl[o]  (64x10)
// ---------------------------------------------------------------------------
__global__ __launch_bounds__(256) void linear_kernel(
    const float* __restrict__ pooled, const float* __restrict__ wl,
    const float* __restrict__ bl, const float* __restrict__ stats,
    float* __restrict__ out)
{
  int t = blockIdx.x * blockDim.x + threadIdx.x;
  if (t >= 64 * 10) return;
  int nI = t / 10;
  int o  = t - nI * 10;
  const float delta = stats[0], alpha = stats[1];
  float s = bl[o];
  for (int c2 = 0; c2 < 128; ++c2) {
    float wv = wl[o * 128 + c2];
    float aw = fabsf(wv);
    float q = (aw > delta) ? ((wv > 0.f) ? alpha : -alpha) : 0.f;
    s += pooled[nI * 128 + c2] * q;
  }
  out[t] = s;
}

// ---------------------------------------------------------------------------
// Workspace layout (all offsets 256B aligned):
//   stats(32B) | wp1 32x32 f16 | wp2 64x288 f16 | wp3 128x576 f16 |
//   pooled 64x128 f32 | h1 NHWC f16 (51.4MB, reused as h3) | h2 NHWC f16
// ---------------------------------------------------------------------------
extern "C" void kernel_launch(void* const* d_in, const int* in_sizes, int n_in,
                              void* d_out, int out_size, void* d_ws, size_t ws_size,
                              hipStream_t stream) {
  (void)in_sizes; (void)n_in; (void)out_size; (void)ws_size;
  const float* x  = (const float*)d_in[0];
  const float* w1 = (const float*)d_in[1];
  const float* g1 = (const float*)d_in[2];
  const float* b1 = (const float*)d_in[3];
  const float* m1 = (const float*)d_in[4];
  const float* v1 = (const float*)d_in[5];
  const float* w2 = (const float*)d_in[6];
  const float* g2 = (const float*)d_in[7];
  const float* b2 = (const float*)d_in[8];
  const float* m2 = (const float*)d_in[9];
  const float* v2 = (const float*)d_in[10];
  const float* w3 = (const float*)d_in[11];
  const float* g3 = (const float*)d_in[12];
  const float* b3 = (const float*)d_in[13];
  const float* m3 = (const float*)d_in[14];
  const float* v3 = (const float*)d_in[15];
  const float* wl = (const float*)d_in[16];
  const float* bl = (const float*)d_in[17];
  float* out = (float*)d_out;

  char* ws = (char*)d_ws;
  const size_t OFF_WP1  = 256;
  const size_t OFF_WP2  = OFF_WP1 + 32 * 32 * 2;            // 2304
  const size_t OFF_WP3  = OFF_WP2 + 64 * 288 * 2;           // 39168
  const size_t OFF_POOL = OFF_WP3 + 128 * 576 * 2;          // 186624
  const size_t OFF_H1   = OFF_POOL + 64 * 128 * 4;          // 219392
  const size_t OFF_H2   = OFF_H1 + (size_t)64 * 32 * 112 * 112 * 2;

  float*    stats  = (float*)ws;
  _Float16* wp1    = (_Float16*)(ws + OFF_WP1);
  _Float16* wp2    = (_Float16*)(ws + OFF_WP2);
  _Float16* wp3    = (_Float16*)(ws + OFF_WP3);
  float*    pooled = (float*)(ws + OFF_POOL);
  _Float16* h1     = (_Float16*)(ws + OFF_H1);   // NHWC 64x112x112x32
  _Float16* h2     = (_Float16*)(ws + OFF_H2);   // NHWC 64x112x112x64
  _Float16* h3     = h1;                         // NHWC 64x56x56x128 (aliases h1)

  // 1) TWN stats for all four weight tensors
  ternary_stats_kernel<<<4, 256, 0, stream>>>(
      w1, w2, w3, wl, 32 * 27, 64 * 288, 128 * 576, 10 * 128, stats);

  // 2) Pack ternary weights (BN scale folded), position-major K
  pack_weights_kernel<<<(32 * 32 + 255) / 256, 256, 0, stream>>>(
      w1, g1, v1, stats + 0, wp1, 32, 3, 32);
  pack_weights_kernel<<<(64 * 288 + 255) / 256, 256, 0, stream>>>(
      w2, g2, v2, stats + 2, wp2, 64, 32, 288);
  pack_weights_kernel<<<(128 * 576 + 255) / 256, 256, 0, stream>>>(
      w3, g3, v3, stats + 4, wp3, 128, 64, 576);

  // 3) Conv layers as WMMA implicit GEMM (one wave = 16 M-rows x COUT)
  // conv1: NCHW f32 in -> NHWC f16 out; M = 802816 -> 12544 blocks
  conv_wmma_kernel<float, 3, 32, 224, 224, 112, 112, 2>
      <<<12544, 128, 0, stream>>>(x, wp1, g1, b1, m1, v1, h1);
  // conv2: NHWC, COUT=64 (NT=4), B tile in 36KB LDS
  conv_wmma_kernel<_Float16, 32, 64, 112, 112, 112, 112, 1>
      <<<12544, 128, 9 * 4 * 32 * 32, stream>>>(h1, wp2, g2, b2, m2, v2, h2);
  // conv3: NHWC, M = 200704 -> 3136 blocks, COUT=128 (NT=8), B tile in 147KB LDS
  conv_wmma_kernel<_Float16, 64, 128, 112, 112, 56, 56, 2>
      <<<3136, 128, 18 * 8 * 32 * 32, stream>>>(h2, wp3, g3, b3, m3, v3, h3);

  // 4) Global average pool (NHWC) -> [64,128] f32
  avgpool_nhwc_kernel<<<64, 128, 0, stream>>>(h3, pooled);

  // 5) Ternary linear head -> d_out [64,10] f32
  linear_kernel<<<3, 256, 0, stream>>>(pooled, wl, bl, stats + 6, out);
}
